// DeltaNetSample_46926812676536
// MI455X (gfx1250) — matrix operand order repaired and checked by reading, not compiled, not run on hardware
//
#include <hip/hip_runtime.h>
#include <math.h>

// ---------------------------------------------------------------------------
// DGA point-cloud network forward for MI455X (gfx1250, wave32).
// All dense linear algebra runs through v_wmma_f32_16x16x32_bf16 (bf16 A/B,
// f32 accumulate).  GEMM A operands have K padded to a multiple of 32
// (zeroed) so the inner loop is unconditional b128 loads + pack + WMMA.
// Waves own 16x32 output tiles (one A fragment feeds two WMMAs) and the
// k-loop is software-pipelined (next loads issued before current WMMAs).
// ---------------------------------------------------------------------------

#define NPTS_ 4096
#define BATCH 4
#define NTOT  (BATCH*NPTS_)   // 16384
#define KG_   16
#define KN_   10
#define EDG   (NTOT*KG_)      // 262144
#define KW_   1.0f
#define REG_  0.001f

typedef __attribute__((ext_vector_type(16))) __bf16 v16bf;
typedef __attribute__((ext_vector_type(16))) float  v16f;
typedef __attribute__((ext_vector_type(8)))  float  v8f;

__device__ __forceinline__ unsigned short f2bf_u(float f) {
  union { float f; unsigned u; } x; x.f = f;
  unsigned u = x.u;
  unsigned r = u + 0x7FFFu + ((u >> 16) & 1u);
  return (unsigned short)(r >> 16);
}

// Pack 16 f32 (four float4) into a 16xbf16 A-fragment (vector fptrunc, RNE).
__device__ __forceinline__ v16bf pack16(float4 f0, float4 f1, float4 f2, float4 f3) {
  v16f t;
  t[0]  = f0.x; t[1]  = f0.y; t[2]  = f0.z; t[3]  = f0.w;
  t[4]  = f1.x; t[5]  = f1.y; t[6]  = f1.z; t[7]  = f1.w;
  t[8]  = f2.x; t[9]  = f2.y; t[10] = f2.z; t[11] = f2.w;
  t[12] = f3.x; t[13] = f3.y; t[14] = f3.z; t[15] = f3.w;
  return __builtin_convertvector(t, v16bf);
}

__device__ __forceinline__ v16bf load_a(const float* __restrict__ arow, int kb, int half) {
  int o0 = kb + half * 8;
  int o1 = kb + 16 + half * 8;
  float4 f0 = *(const float4*)(arow + o0);
  float4 f1 = *(const float4*)(arow + o0 + 4);
  float4 f2 = *(const float4*)(arow + o1);
  float4 f3 = *(const float4*)(arow + o1 + 4);
  return pack16(f0, f1, f2, f3);
}

__device__ __forceinline__ v16bf load_b(const unsigned short* __restrict__ wrow, int kw) {
  union { uint4 q[2]; v16bf v; } u;
  int kt = (kw & ~16) + ((kw & 16) >> 1);
  u.q[0] = *(const uint4*)(wrow + kt);
  u.q[1] = *(const uint4*)(wrow + kt + 16);
  return u.v;
}

__device__ __forceinline__ v8f wmma_bf16(v16bf a, v16bf b, v8f c) {
  return __builtin_amdgcn_wmma_f32_16x16x32_bf16(false, a, false, b, (short)0, c, false, false);
}

// ---------------------------------------------------------------------------
// Weight f32 -> bf16 conversion with K padded to a multiple of 32
// ---------------------------------------------------------------------------
__global__ void cvt_bf16_kernel(unsigned short* __restrict__ dst,
                                const float* __restrict__ src,
                                int rows, int K, int Kp) {
  int t = blockIdx.x * 256 + threadIdx.x;
  if (t >= rows * Kp) return;
  int r = t / Kp, k = t % Kp;
  dst[t] = (k < K) ? f2bf_u(src[(size_t)r * K + k]) : (unsigned short)0;
}

// ---------------------------------------------------------------------------
// Generic GEMM: out[M,N] = act((A[M,lda]*W[N,Kp]^T + bias)*scale*smul + shift) (+resAdd)
// 16x32 output tile per wave; pipelined k-loop.  act: 0 none, 1 relu, 2 sigmoid.
// ---------------------------------------------------------------------------
__global__ void gemm_kernel(const float* __restrict__ A, int lda,
                            const unsigned short* __restrict__ Wb,
                            const float* __restrict__ bias,
                            const float* __restrict__ scale,
                            const float* __restrict__ shift,
                            const float* __restrict__ resAdd,
                            float* __restrict__ out,
                            int M, int K, int Kp, int Nout, int act, float smul) {
  int wave = (blockIdx.x * blockDim.x + threadIdx.x) >> 5;
  int lane = threadIdx.x & 31;
  int ntn = Nout >> 5;                         // 32-column tiles
  int total = (M >> 4) * ntn;
  if (wave >= total) return;
  int tm = wave / ntn, tn = wave % ntn;
  int row = lane & 15, half = lane >> 4;
  int m  = tm * 16 + row;
  int n0 = tn * 32 + row, n1 = n0 + 16;
  const float* arow = A + (size_t)m * lda;
  const unsigned short* w0 = Wb + (size_t)n0 * Kp;
  const unsigned short* w1 = Wb + (size_t)n1 * Kp;
  v8f c0 = {}, c1 = {};
  v16bf a  = load_a(arow, 0, half);
  v16bf b0 = load_b(w0, half * 16);
  v16bf b1 = load_b(w1, half * 16);
  for (int kb = 32; kb < K; kb += 32) {
    v16bf an  = load_a(arow, kb, half);
    v16bf b0n = load_b(w0, kb + half * 16);
    v16bf b1n = load_b(w1, kb + half * 16);
    c0 = wmma_bf16(a, b0, c0);
    c1 = wmma_bf16(a, b1, c1);
    a = an; b0 = b0n; b1 = b1n;
  }
  c0 = wmma_bf16(a, b0, c0);
  c1 = wmma_bf16(a, b1, c1);
  int   ncol[2] = { n0, n1 };
  v8f   cc[2];  cc[0] = c0;  cc[1] = c1;
#pragma unroll
  for (int t = 0; t < 2; t++) {
    int n = ncol[t];
    float bv = bias ? bias[n] : 0.f;
    float ss = (scale ? scale[n] : 1.f) * smul;
    float sh = shift ? shift[n] : 0.f;
#pragma unroll
    for (int j = 0; j < 8; j++) {
      int mm = tm * 16 + j + 8 * half;
      float y = (cc[t][j] + bv) * ss + sh;
      if (act == 1)      y = fmaxf(y, 0.f);
      else if (act == 2) y = 1.f / (1.f + expf(-y));
      if (resAdd) y += resAdd[(size_t)mm * Nout + n];
      out[(size_t)mm * Nout + n] = y;
    }
  }
}

// ---------------------------------------------------------------------------
// Fused MultiChannelGLU: out = sum_c softmax(alpha)_c * (A Wv_c^T + bv_c)
//                              * sigmoid(A Wg_c^T + bg_c),  K == C (square).
// All 4 value + 4 gate channels accumulate in one pass: each A fragment is
// loaded once and feeds 8 WMMAs.
// ---------------------------------------------------------------------------
__global__ void glu_kernel(const float* __restrict__ A, int lda,
                           const unsigned short* __restrict__ Wv,
                           const unsigned short* __restrict__ Wg, int Kp,
                           const float* __restrict__ bv, const float* __restrict__ bg,
                           const float* __restrict__ alpha,
                           float* __restrict__ out, int M, int C) {
  int wave = (blockIdx.x * blockDim.x + threadIdx.x) >> 5;
  int lane = threadIdx.x & 31;
  int ntn = C >> 4;
  int total = (M >> 4) * ntn;
  if (wave >= total) return;
  int tm = wave / ntn, tn = wave % ntn;
  int row = lane & 15, half = lane >> 4;
  int m = tm * 16 + row;
  int n = tn * 16 + row;
  const float* arow = A + (size_t)m * lda;
  v8f z = {};
  v8f aV[4], aG[4];
#pragma unroll
  for (int ch = 0; ch < 4; ch++) { aV[ch] = z; aG[ch] = z; }
  for (int kb = 0; kb < C; kb += 32) {
    v16bf a = load_a(arow, kb, half);
    int kw = kb + half * 16;
#pragma unroll
    for (int ch = 0; ch < 4; ch++) {
      aV[ch] = wmma_bf16(a, load_b(Wv + ((size_t)ch * C + n) * Kp, kw), aV[ch]);
      aG[ch] = wmma_bf16(a, load_b(Wg + ((size_t)ch * C + n) * Kp, kw), aG[ch]);
    }
  }
  float a0 = alpha[0], a1 = alpha[1], a2 = alpha[2], a3 = alpha[3];
  float mx = fmaxf(fmaxf(a0, a1), fmaxf(a2, a3));
  float e0 = expf(a0 - mx), e1 = expf(a1 - mx), e2 = expf(a2 - mx), e3 = expf(a3 - mx);
  float ies = 1.f / (e0 + e1 + e2 + e3);
  float wch[4] = { e0 * ies, e1 * ies, e2 * ies, e3 * ies };
  float res[8] = { 0, 0, 0, 0, 0, 0, 0, 0 };
#pragma unroll
  for (int ch = 0; ch < 4; ch++) {
    float bvc = bv[ch * C + n], bgc = bg[ch * C + n];
    float w = wch[ch];
#pragma unroll
    for (int j = 0; j < 8; j++) {
      float g = 1.f / (1.f + expf(-(aG[ch][j] + bgc)));
      res[j] += w * (aV[ch][j] + bvc) * g;
    }
  }
#pragma unroll
  for (int j = 0; j < 8; j++)
    out[(size_t)(tm * 16 + j + 8 * half) * C + n] = res[j];
}

// ---------------------------------------------------------------------------
// Fused per-edge GEMM + k-reduction.  Tile = the 16 edges of point i.
// mode 0: a[k] = X[nbr[e]][k] - cen*X[i][k]     (s_mlp_max; relu∘max commute)
// mode 1: a[k] = X[i][k] * F2[e][k]             (post stage; relu then sum)
// reduceSum: 0 -> max over the 16 edges, 1 -> sum.
// transposeOut: store as out[b][col][n] (final output layout).
// ---------------------------------------------------------------------------
__device__ __forceinline__ v16bf edge_load_a(const float* __restrict__ grow,
                                             const float* __restrict__ crow,
                                             int kb, int half, int mode, int cen) {
  int o0 = kb + half * 8;
  int o1 = kb + 16 + half * 8;
  float4 g0 = *(const float4*)(grow + o0);
  float4 g1 = *(const float4*)(grow + o0 + 4);
  float4 g2 = *(const float4*)(grow + o1);
  float4 g3 = *(const float4*)(grow + o1 + 4);
  if (mode == 0) {
    if (cen) {
      float4 c0 = *(const float4*)(crow + o0);
      float4 c1 = *(const float4*)(crow + o0 + 4);
      float4 c2 = *(const float4*)(crow + o1);
      float4 c3 = *(const float4*)(crow + o1 + 4);
      g0.x -= c0.x; g0.y -= c0.y; g0.z -= c0.z; g0.w -= c0.w;
      g1.x -= c1.x; g1.y -= c1.y; g1.z -= c1.z; g1.w -= c1.w;
      g2.x -= c2.x; g2.y -= c2.y; g2.z -= c2.z; g2.w -= c2.w;
      g3.x -= c3.x; g3.y -= c3.y; g3.z -= c3.z; g3.w -= c3.w;
    }
  } else {
    float4 c0 = *(const float4*)(crow + o0);
    float4 c1 = *(const float4*)(crow + o0 + 4);
    float4 c2 = *(const float4*)(crow + o1);
    float4 c3 = *(const float4*)(crow + o1 + 4);
    g0.x *= c0.x; g0.y *= c0.y; g0.z *= c0.z; g0.w *= c0.w;
    g1.x *= c1.x; g1.y *= c1.y; g1.z *= c1.z; g1.w *= c1.w;
    g2.x *= c2.x; g2.y *= c2.y; g2.z *= c2.z; g2.w *= c2.w;
    g3.x *= c3.x; g3.y *= c3.y; g3.z *= c3.z; g3.w *= c3.w;
  }
  return pack16(g0, g1, g2, g3);
}

__global__ void edge_gemm_reduce_kernel(const float* __restrict__ X, int ldx,
                                        const float* __restrict__ F2,
                                        const int* __restrict__ nbr,
                                        const unsigned short* __restrict__ Wb,
                                        const float* __restrict__ bias,
                                        const float* __restrict__ scale,
                                        const float* __restrict__ shift,
                                        float* __restrict__ out,
                                        int K, int Kp, int Nout,
                                        int mode, int cen, int reduceSum,
                                        int act, float smul, int transposeOut) {
  int wave = (blockIdx.x * blockDim.x + threadIdx.x) >> 5;
  int lane = threadIdx.x & 31;
  int ntn = Nout >> 5;                          // 32-column tiles
  int total = NTOT * ntn;
  if (wave >= total) return;
  int i  = wave / ntn;
  int tn = wave % ntn;
  int row = lane & 15, half = lane >> 4;
  int e  = i * KG_ + row;                       // edge handled by this lane (A row)
  int nb = nbr[e];
  const float* crow = X + (size_t)i * ldx;
  const float* grow = (mode == 0) ? (X + (size_t)nb * ldx) : (F2 + (size_t)e * K);
  int n0 = tn * 32 + row, n1 = n0 + 16;
  const unsigned short* w0 = Wb + (size_t)n0 * Kp;
  const unsigned short* w1 = Wb + (size_t)n1 * Kp;
  v8f c0 = {}, c1 = {};
  v16bf a  = edge_load_a(grow, crow, 0, half, mode, cen);
  v16bf b0 = load_b(w0, half * 16);
  v16bf b1 = load_b(w1, half * 16);
  for (int kb = 32; kb < K; kb += 32) {
    v16bf an  = edge_load_a(grow, crow, kb, half, mode, cen);
    v16bf b0n = load_b(w0, kb + half * 16);
    v16bf b1n = load_b(w1, kb + half * 16);
    c0 = wmma_bf16(a, b0, c0);
    c1 = wmma_bf16(a, b1, c1);
    a = an; b0 = b0n; b1 = b1n;
  }
  c0 = wmma_bf16(a, b0, c0);
  c1 = wmma_bf16(a, b1, c1);
  int ncol[2] = { n0, n1 };
  v8f cc[2];  cc[0] = c0;  cc[1] = c1;
#pragma unroll
  for (int t = 0; t < 2; t++) {
    int n = ncol[t];
    float bv = bias ? bias[n] : 0.f;
    float ss = (scale ? scale[n] : 1.f) * smul;
    float sh = shift ? shift[n] : 0.f;
    float red = reduceSum ? 0.f : -3.4e38f;
#pragma unroll
    for (int j = 0; j < 8; j++) {
      float y = (cc[t][j] + bv) * ss + sh;
      if (act == 1)      y = fmaxf(y, 0.f);
      else if (act == 2) y = 1.f / (1.f + expf(-y));
      red = reduceSum ? (red + y) : fmaxf(red, y);
    }
    float other = __shfl_xor(red, 16, 32);
    red = reduceSum ? (red + other) : fmaxf(red, other);
    if (half == 0) {
      if (transposeOut) {
        int b_ = i / NPTS_, pn = i % NPTS_;
        out[((size_t)b_ * Nout + n) * NPTS_ + pn] = red;
      } else {
        out[(size_t)i * Nout + n] = red;
      }
    }
  }
}

// ---------------------------------------------------------------------------
// Preprocessing kernels
// ---------------------------------------------------------------------------
__global__ void transpose_pts_kernel(float* __restrict__ pos, float* __restrict__ posp,
                                     const float* __restrict__ pts) {
  int i = blockIdx.x * 256 + threadIdx.x;
  if (i >= NTOT) return;
  int b = i / NPTS_, n = i % NPTS_;
#pragma unroll
  for (int d = 0; d < 3; d++) {
    float v = pts[((size_t)b * 3 + d) * NPTS_ + n];
    pos[(size_t)i * 3 + d]  = v;
    posp[(size_t)i * 32 + d] = v;   // padded copy for WMMA consumption
  }
}

__global__ void knn_kernel(const float* __restrict__ pos, int* __restrict__ nbr) {
  __shared__ float sx[256], sy[256], sz[256];
  int b = blockIdx.y;
  int q = blockIdx.x * 256 + threadIdx.x;
  int gi = b * NPTS_ + q;
  float px = pos[(size_t)gi * 3], py = pos[(size_t)gi * 3 + 1], pz = pos[(size_t)gi * 3 + 2];
  float bd[KG_]; int bi_[KG_];
#pragma unroll
  for (int k = 0; k < KG_; k++) { bd[k] = 3.4e38f; bi_[k] = 0; }
  for (int ch = 0; ch < NPTS_ / 256; ch++) {
    int gc = b * NPTS_ + ch * 256 + threadIdx.x;
    sx[threadIdx.x] = pos[(size_t)gc * 3];
    sy[threadIdx.x] = pos[(size_t)gc * 3 + 1];
    sz[threadIdx.x] = pos[(size_t)gc * 3 + 2];
    if (ch + 1 < NPTS_ / 256)
      __builtin_prefetch(&pos[(size_t)(gc + 256) * 3], 0, 1);   // global_prefetch_b8
    __syncthreads();
    for (int t = 0; t < 256; t++) {
      float dx = sx[t] - px, dy = sy[t] - py, dz = sz[t] - pz;
      float d2 = dx * dx + dy * dy + dz * dz;
      if (d2 < bd[KG_ - 1]) {
        int idx = ch * 256 + t;
        int j = KG_ - 1;
        while (j > 0 && bd[j - 1] > d2) { bd[j] = bd[j - 1]; bi_[j] = bi_[j - 1]; j--; }
        bd[j] = d2; bi_[j] = idx;
      }
    }
    __syncthreads();
  }
#pragma unroll
  for (int k = 0; k < KG_; k++) nbr[(size_t)gi * KG_ + k] = b * NPTS_ + bi_[k];
}

// Tangent basis (smallest eigvec of 10-NN covariance) + MLS gradient coeffs.
__global__ void basis_grad_kernel(const float* __restrict__ pos,
                                  const int* __restrict__ nbr,
                                  float* __restrict__ gcoef) {
  int i = blockIdx.x * blockDim.x + threadIdx.x;
  if (i >= NTOT) return;
  float px = pos[(size_t)i * 3], py = pos[(size_t)i * 3 + 1], pz = pos[(size_t)i * 3 + 2];
  float c00 = 0, c01 = 0, c02 = 0, c11 = 0, c12 = 0, c22 = 0;
#pragma unroll
  for (int k = 0; k < KN_; k++) {
    int nb = nbr[(size_t)i * KG_ + k];
    float lx = pos[(size_t)nb * 3] - px, ly = pos[(size_t)nb * 3 + 1] - py, lz = pos[(size_t)nb * 3 + 2] - pz;
    c00 += lx * lx; c01 += lx * ly; c02 += lx * lz;
    c11 += ly * ly; c12 += ly * lz; c22 += lz * lz;
  }
  float q  = (c00 + c11 + c22) / 3.f;
  float p1 = c01 * c01 + c02 * c02 + c12 * c12;
  float a00 = c00 - q, a11 = c11 - q, a22 = c22 - q;
  float p2 = a00 * a00 + a11 * a11 + a22 * a22 + 2.f * p1;
  float p  = sqrtf(p2 / 6.f + 1e-30f);
  float ip = 1.f / p;
  float b00 = a00 * ip, b01 = c01 * ip, b02 = c02 * ip;
  float b11 = a11 * ip, b12 = c12 * ip, b22 = a22 * ip;
  float detB = b00 * (b11 * b22 - b12 * b12) - b01 * (b01 * b22 - b12 * b02)
             + b02 * (b01 * b12 - b11 * b02);
  float r = fminf(1.f, fmaxf(-1.f, 0.5f * detB));
  float phi = acosf(r) / 3.f;
  float e3 = q + 2.f * p * cosf(phi + 2.0943951023931953f);
  float r0x = c00 - e3, r0y = c01,     r0z = c02;
  float r1x = c01,     r1y = c11 - e3, r1z = c12;
  float r2x = c02,     r2y = c12,     r2z = c22 - e3;
  float v1x = r0y * r1z - r0z * r1y, v1y = r0z * r1x - r0x * r1z, v1z = r0x * r1y - r0y * r1x;
  float v2x = r0y * r2z - r0z * r2y, v2y = r0z * r2x - r0x * r2z, v2z = r0x * r2y - r0y * r2x;
  float v3x = r1y * r2z - r1z * r2y, v3y = r1z * r2x - r1x * r2z, v3z = r1x * r2y - r1y * r2x;
  float n1 = v1x * v1x + v1y * v1y + v1z * v1z;
  float n2 = v2x * v2x + v2y * v2y + v2z * v2z;
  float n3 = v3x * v3x + v3y * v3y + v3z * v3z;
  float nx = v1x, ny = v1y, nz = v1z, nn = n1;
  if (n2 > nn) { nx = v2x; ny = v2y; nz = v2z; nn = n2; }
  if (n3 > nn) { nx = v3x; ny = v3y; nz = v3z; nn = n3; }
  float inn = rsqrtf(nn + 1e-30f);
  nx *= inn; ny *= inn; nz *= inn;
  if (nx * px + ny * py + nz * pz < 0.f) { nx = -nx; ny = -ny; nz = -nz; }
  float tx, ty, tz;
  if (fabsf(nx) > 0.9f) { tx = 0.f; ty = 1.f; tz = 0.f; } else { tx = 1.f; ty = 0.f; tz = 0.f; }
  float xbx = ty * nz - tz * ny, xby = tz * nx - tx * nz, xbz = tx * ny - ty * nx;
  float xl = 1.f / (sqrtf(xbx * xbx + xby * xby + xbz * xbz) + 1e-12f);
  xbx *= xl; xby *= xl; xbz *= xl;
  float ybx = ny * xbz - nz * xby, yby = nz * xbx - nx * xbz, ybz = nx * xby - ny * xbx;
  float u[KG_], w[KG_], dd[KG_];
  float msum = 0.f;
#pragma unroll
  for (int k = 0; k < KG_; k++) {
    int nb = nbr[(size_t)i * KG_ + k];
    float lx = pos[(size_t)nb * 3] - px, ly = pos[(size_t)nb * 3 + 1] - py, lz = pos[(size_t)nb * 3 + 2] - pz;
    u[k]  = lx * xbx + ly * xby + lz * xbz;
    w[k]  = lx * ybx + ly * yby + lz * ybz;
    dd[k] = lx * lx + ly * ly + lz * lz;
    msum += sqrtf(dd[k] + 1e-12f);
  }
  float sigma = KW_ * (msum / KG_) + 1e-12f;
  float is2 = 1.f / (sigma * sigma);
  float m00 = REG_, m01 = 0, m02 = 0, m11 = REG_, m12 = 0, m22 = REG_;
  float wt[KG_];
#pragma unroll
  for (int k = 0; k < KG_; k++) {
    float t = expf(-dd[k] * is2);
    wt[k] = t;
    m00 += t;               m01 += t * u[k];        m02 += t * w[k];
    m11 += t * u[k] * u[k]; m12 += t * u[k] * w[k]; m22 += t * w[k] * w[k];
  }
  float det = m00 * (m11 * m22 - m12 * m12) - m01 * (m01 * m22 - m12 * m02)
            + m02 * (m01 * m12 - m11 * m02);
  float id = 1.f / det;
  float i01 = (m02 * m12 - m01 * m22) * id;
  float i11 = (m00 * m22 - m02 * m02) * id;
  float i12 = (m01 * m02 - m00 * m12) * id;
  float i02 = (m01 * m12 - m02 * m11) * id;
  float i22 = (m00 * m11 - m01 * m01) * id;
#pragma unroll
  for (int k = 0; k < KG_; k++) {
    gcoef[((size_t)i * 2 + 0) * KG_ + k] = wt[k] * (i01 + i11 * u[k] + i12 * w[k]);
    gcoef[((size_t)i * 2 + 1) * KG_ + k] = wt[k] * (i02 + i12 * u[k] + i22 * w[k]);
  }
}

// ---------------------------------------------------------------------------
// grad / div operators, concats, epilogues
// ---------------------------------------------------------------------------
__global__ void grad_apply_kernel(float* __restrict__ out, const float* __restrict__ X,
                                  const float* __restrict__ g, const int* __restrict__ nbr, int C) {
  int t = blockIdx.x * 256 + threadIdx.x;
  if (t >= NTOT * C) return;
  int i = t / C, ch = t % C;
  const float* g0 = g + (size_t)i * 2 * KG_;
  const int* nb = nbr + (size_t)i * KG_;
  float a0 = 0.f, a1 = 0.f;
#pragma unroll
  for (int k = 0; k < KG_; k++) {
    float xx = X[(size_t)nb[k] * C + ch];
    a0 += g0[k] * xx;
    a1 += g0[KG_ + k] * xx;
  }
  out[((size_t)i * 2 + 0) * C + ch] = a0;
  out[((size_t)i * 2 + 1) * C + ch] = a1;
}

__global__ void div_apply_kernel(float* __restrict__ out, const float* __restrict__ v,
                                 const float* __restrict__ g, const int* __restrict__ nbr,
                                 int C, int jflag) {
  int t = blockIdx.x * 256 + threadIdx.x;
  if (t >= NTOT * C) return;
  int i = t / C, ch = t % C;
  float v0 = v[((size_t)i * 2 + 0) * C + ch];
  float v1 = v[((size_t)i * 2 + 1) * C + ch];
  float a0 = jflag ? -v1 : v0;
  float a1 = jflag ?  v0 : v1;
  const float* g0 = g + (size_t)i * 2 * KG_;
  const int* nb = nbr + (size_t)i * KG_;
#pragma unroll
  for (int k = 0; k < KG_; k++) {
    float contrib = -(g0[k] * a0 + g0[KG_ + k] * a1);
    atomicAdd(&out[(size_t)nb[k] * C + ch], contrib);
  }
}

__global__ void cat_x_kernel(float* __restrict__ xc, int ldo, const float* __restrict__ x,
                             const float* __restrict__ d1, const float* __restrict__ d2,
                             const float* __restrict__ v, int cin) {
  int t = blockIdx.x * 256 + threadIdx.x;
  int tot = NTOT * 4 * cin;
  if (t >= tot) return;
  int i = t / (4 * cin), col = t % (4 * cin);
  float val;
  if (col < cin)            val = x[(size_t)i * cin + col];
  else if (col < 2 * cin)   val = d1[(size_t)i * cin + col - cin];
  else if (col < 3 * cin)   val = d2[(size_t)i * cin + col - 2 * cin];
  else {
    int ch = col - 3 * cin;
    float v0 = v[((size_t)i * 2 + 0) * cin + ch];
    float v1 = v[((size_t)i * 2 + 1) * cin + ch];
    val = sqrtf(v0 * v0 + v1 * v1 + 1e-12f);
  }
  xc[(size_t)i * ldo + col] = val;
}

__global__ void vcat_kernel(float* __restrict__ vc, int ldo, const float* __restrict__ v,
                            const float* __restrict__ g1, const float* __restrict__ g2,
                            const float* __restrict__ gx, int cin, int cout) {
  int Kv = 2 * cin + cout;
  int t = blockIdx.x * 256 + threadIdx.x;
  if (t >= NTOT * Kv) return;
  int i = t / Kv, col = t % Kv;
  float val0, val1;
  if (col < cin) {
    val0 = v[((size_t)i * 2 + 0) * cin + col];
    val1 = v[((size_t)i * 2 + 1) * cin + col];
  } else if (col < 2 * cin) {
    int ch = col - cin;
    float a0 = g1[((size_t)i * 2 + 0) * cin + ch], a1 = g1[((size_t)i * 2 + 1) * cin + ch];
    float b0 = g2[((size_t)i * 2 + 0) * cin + ch], b1 = g2[((size_t)i * 2 + 1) * cin + ch];
    val0 = -a0 + b1;           // hl = -g1 - J(g2)
    val1 = -a1 - b0;
  } else {
    int ch = col - 2 * cin;
    val0 = gx[((size_t)i * 2 + 0) * cout + ch];
    val1 = gx[((size_t)i * 2 + 1) * cout + ch];
  }
  float* r0 = vc + ((size_t)i * 2 + 0) * ldo;
  float* r1 = vc + ((size_t)i * 2 + 1) * ldo;
  r0[col] = val0;        r1[col] = val1;
  r0[Kv + col] = -val1;  r1[Kv + col] = val0;   // I_J doubling
}

__global__ void ln_kernel(float* __restrict__ out, const float* __restrict__ in,
                          const float* __restrict__ g, const float* __restrict__ b, int C) {
  int i = blockIdx.x * 256 + threadIdx.x;
  if (i >= NTOT) return;
  const float* r = in + (size_t)i * C;
  float mx = -3.4e38f;
  for (int c = 0; c < C; c++) mx = fmaxf(mx, r[c]);
  float s = 0.f;
  for (int c = 0; c < C; c++) s += r[c] + mx;
  float m = s / (float)C;
  float var = 0.f;
  for (int c = 0; c < C; c++) { float d = r[c] + mx - m; var += d * d; }
  var /= (float)C;
  float inv = rsqrtf(var + 1e-5f);
  float* o = out + (size_t)i * C;
  for (int c = 0; c < C; c++) o[c] = (r[c] + mx - m) * inv * g[c] + b[c];
}

__global__ void vmean_kernel(float* __restrict__ out, const float* __restrict__ in, int C) {
  int r = blockIdx.x * 256 + threadIdx.x;
  if (r >= 2 * NTOT) return;
  const float* p = in + (size_t)r * C;
  float s = 0.f;
  for (int c = 0; c < C; c++) s += p[c];
  s /= (float)C;
  float* o = out + (size_t)r * C;
  for (int c = 0; c < C; c++) o[c] = p[c] + s;
}

// ---------------------------------------------------------------------------
// SE block + delta prep
// ---------------------------------------------------------------------------
__global__ void se_reduce_kernel(float* __restrict__ s, const float* __restrict__ x) {
  int t = blockIdx.x * 256 + threadIdx.x;
  if (t >= BATCH * 128) return;
  int b = t / 128, c = t % 128;
  float acc = 0.f;
  for (int n = 0; n < NPTS_; n++) acc += x[((size_t)b * NPTS_ + n) * 128 + c];
  s[t] = acc / (float)NPTS_;
}

__global__ void se_mlp_kernel(float* __restrict__ sfac, const float* __restrict__ s,
                              const float* __restrict__ w1, const float* __restrict__ b1,
                              const float* __restrict__ g1, const float* __restrict__ be1,
                              const float* __restrict__ w2, const float* __restrict__ b2,
                              const float* __restrict__ g2, const float* __restrict__ be2) {
  __shared__ float h[BATCH * 4];
  int t = threadIdx.x;
  const float bninv = rsqrtf(1.0f + 1e-5f);
  if (t < BATCH * 4) {
    int b = t / 4, r = t % 4;
    float acc = b1[r];
    for (int k = 0; k < 128; k++) acc += w1[r * 128 + k] * s[b * 128 + k];
    acc = acc * bninv * g1[r] + be1[r];
    h[t] = fmaxf(acc, 0.f);
  }
  __syncthreads();
  if (t < BATCH * 128) {
    int b = t / 128, c = t % 128;
    float acc = b2[c];
    for (int k = 0; k < 4; k++) acc += w2[c * 4 + k] * h[b * 4 + k];
    acc = acc * bninv * g2[c] + be2[c];
    sfac[t] = 1.f / (1.f + expf(-acc));
  }
}

__global__ void se_apply_kernel(float* __restrict__ f, const float* __restrict__ x,
                                const float* __restrict__ sfac) {
  int t = blockIdx.x * 256 + threadIdx.x;
  if (t >= NTOT * 128) return;
  int i = t / 128, c = t % 128;
  int b = i / NPTS_;
  f[t] = x[t] * sfac[b * 128 + c];
}

__global__ void kd3_kernel(float* __restrict__ kd, const float* __restrict__ pos,
                           const int* __restrict__ nbr) {
  int e = blockIdx.x * 256 + threadIdx.x;
  if (e >= EDG) return;
  int i = e / KG_;
  int nb = nbr[e];
#pragma unroll
  for (int j = 0; j < 3; j++) {
    float d = pos[(size_t)nb * 3 + j] - pos[(size_t)i * 3 + j];
    kd[(size_t)e * 32 + j] = d + sinf(d) * cosf(d);   // ld = 32 (padded)
  }
}

// ---------------------------------------------------------------------------
// Host orchestration
// ---------------------------------------------------------------------------
struct WT { const unsigned short* p; int Kp; };

struct ConvIdx {
  int aso_b, aso_w, asv_b, asv_w, avo_b, avo_w, avv_b, avv_w;
  int gs_alpha, gs_bg, gs_bv, gs_wg, gs_wv;
  int gv_alpha, gv_bg, gv_bv, gv_wg, gv_wv;
  int ln_b, ln_g, smlp_b, smlp_w, smax_b, smax_w, vmlp_b, vmlp_w;
};

// JAX pytree order: top-level dict order {feats, pts, params}; nested dicts
// flattened with sorted keys, lists in order.
static const ConvIdx convIdx[3] = {
  {  2, 3, 4, 5, 6, 7, 8, 9, 10,11,12,13,14, 15,16,17,18,19, 20,21, 22,23, 24,25, 26,27 },
  { 28,29,30,31,32,33,34,35, 36,37,38,39,40, 41,42,43,44,45, 46,47, 48,49, 50,51, 52,53 },
  { 54,55,56,57,58,59,60,61, 62,63,64,65,66, 67,68,69,70,71, 72,73, 74,75, 76,77, -1,-1 },
};

extern "C" void kernel_launch(void* const* d_in, const int* in_sizes, int n_in,
                              void* d_out, int out_size, void* d_ws, size_t ws_size,
                              hipStream_t stream) {
  (void)in_sizes; (void)n_in; (void)out_size; (void)ws_size;
  const float* pts = (const float*)d_in[1];
  char* ws = (char*)d_ws;

  // ---- workspace layout (~320 MB) ----
  size_t o = 0;
  auto alloc = [&](size_t bytes) { size_t r = (o + 255) & ~(size_t)255; o = r + bytes; return r; };
  size_t oNbr  = alloc((size_t)EDG * sizeof(int));
  size_t oPos  = alloc((size_t)NTOT * 3 * 4);
  size_t oPosP = alloc((size_t)NTOT * 32 * 4);       // padded pos for WMMA
  size_t oG    = alloc((size_t)NTOT * 2 * KG_ * 4);
  size_t oX    = alloc((size_t)NTOT * 128 * 4);
  size_t oT1   = alloc((size_t)NTOT * 128 * 4);
  size_t oT2   = alloc((size_t)NTOT * 128 * 4);
  size_t oXC   = alloc((size_t)NTOT * 512 * 4);
  size_t oV    = alloc((size_t)NTOT * 2 * 128 * 4);
  size_t oVT1  = alloc((size_t)NTOT * 2 * 128 * 4);
  size_t oVT2  = alloc((size_t)NTOT * 2 * 128 * 4);
  size_t oGX   = alloc((size_t)NTOT * 2 * 128 * 4);
  size_t oD1   = alloc((size_t)NTOT * 128 * 4);
  size_t oD2   = alloc((size_t)NTOT * 128 * 4);
  size_t oBIG  = alloc((size_t)EDG * 128 * 4);       // vcat during convs / kd128 in post
  size_t oKD3  = alloc((size_t)EDG * 32 * 4);        // padded kd3
  size_t oSE   = alloc(4096);
  size_t oWB   = alloc((size_t)16 << 20);            // bf16 weights

  int*   Nbr  = (int*)(ws + oNbr);
  float* Pos  = (float*)(ws + oPos);
  float* PosP = (float*)(ws + oPosP);
  float* Gc   = (float*)(ws + oG);
  float* X    = (float*)(ws + oX);
  float* T1   = (float*)(ws + oT1);
  float* T2   = (float*)(ws + oT2);
  float* XC   = (float*)(ws + oXC);
  float* V    = (float*)(ws + oV);
  float* VT1  = (float*)(ws + oVT1);
  float* VT2  = (float*)(ws + oVT2);
  float* GX   = (float*)(ws + oGX);
  float* D1   = (float*)(ws + oD1);
  float* D2   = (float*)(ws + oD2);
  float* BIG  = (float*)(ws + oBIG);
  float* KD3  = (float*)(ws + oKD3);
  float* SEs  = (float*)(ws + oSE);
  float* SEf  = SEs + BATCH * 128;

  auto P = [&](int idx) { return (const float*)d_in[idx]; };

  // ---- weight conversion to padded bf16 ----
  size_t wbOff = 0;
  auto cvt = [&](int idx, int rows, int K) -> WT {
    int Kp = (K + 31) & ~31;
    unsigned short* dst = (unsigned short*)(ws + oWB + wbOff);
    wbOff = (wbOff + (size_t)rows * Kp * 2 + 255) & ~(size_t)255;
    int tot = rows * Kp;
    cvt_bf16_kernel<<<(tot + 255) / 256, 256, 0, stream>>>(dst, P(idx), rows, K, Kp);
    return WT{dst, Kp};
  };

  const int cinL[3]  = { 3, 64, 128 };
  const int coutL[3] = { 64, 128, 128 };
  WT Wsmax[3], Wsmlp[3], Wasv[3], Waso[3], Wavv[3], Wavo[3];
  WT Wgsv[3], Wgsg[3], Wgvv[3], Wgvg[3], Wvmlp[2];
  for (int l = 0; l < 3; l++) {
    const ConvIdx& C = convIdx[l];
    Wsmax[l] = cvt(C.smax_w, coutL[l], cinL[l]);
    Wsmlp[l] = cvt(C.smlp_w, coutL[l], 4 * cinL[l]);
    Wasv[l]  = cvt(C.asv_w,  coutL[l], coutL[l]);
    Waso[l]  = cvt(C.aso_w,  coutL[l], coutL[l]);
    Wavv[l]  = cvt(C.avv_w,  coutL[l], coutL[l]);
    Wavo[l]  = cvt(C.avo_w,  coutL[l], coutL[l]);
    Wgsv[l]  = cvt(C.gs_wv, 4 * coutL[l], coutL[l]);
    Wgsg[l]  = cvt(C.gs_wg, 4 * coutL[l], coutL[l]);
    Wgvv[l]  = cvt(C.gv_wv, 4 * coutL[l], coutL[l]);
    Wgvg[l]  = cvt(C.gv_wg, 4 * coutL[l], coutL[l]);
    if (l < 2) Wvmlp[l] = cvt(C.vmlp_w, coutL[l], 2 * (2 * cinL[l] + coutL[l]));
  }
  WT Wdelta = cvt(81, 128, 3);
  WT Wpost  = cvt(85, 128, 128);

  // ---- launch helpers ----
  auto gemm = [&](const float* A, int lda, WT w, const float* bias, const float* scale,
                  const float* shift, const float* resAdd, float* out,
                  int M, int K, int Nout, int act, float smul) {
    int waves = (M / 16) * (Nout / 32);
    gemm_kernel<<<(waves + 7) / 8, 256, 0, stream>>>(A, lda, w.p, bias, scale, shift, resAdd,
                                                     out, M, K, w.Kp, Nout, act, smul);
  };
  auto edge = [&](const float* Xp, int ldx, const float* F2, WT w, const float* bias,
                  const float* scale, const float* shift, float* out, int K, int Nout,
                  int mode, int cen, int rsum, int act, float smul, int transp) {
    int waves = NTOT * (Nout / 32);
    edge_gemm_reduce_kernel<<<(waves + 7) / 8, 256, 0, stream>>>(
        Xp, ldx, F2, Nbr, w.p, bias, scale, shift, out, K, w.Kp, Nout,
        mode, cen, rsum, act, smul, transp);
  };
  auto glu = [&](const float* A, int lda, WT wv, WT wg, const float* bv, const float* bg,
                 const float* alpha, float* out, int M, int C) {
    int waves = (M / 16) * (C / 16);
    glu_kernel<<<(waves + 7) / 8, 256, 0, stream>>>(A, lda, wv.p, wg.p, wv.Kp,
                                                    bv, bg, alpha, out, M, C);
  };
  auto divop = [&](float* dst, const float* v, int C, int jflag) {
    hipMemsetAsync(dst, 0, (size_t)NTOT * C * 4, stream);
    div_apply_kernel<<<((NTOT * C) + 255) / 256, 256, 0, stream>>>(dst, v, Gc, Nbr, C, jflag);
  };

  // ---- preprocessing ----
  hipMemsetAsync(PosP, 0, (size_t)NTOT * 32 * 4, stream);
  transpose_pts_kernel<<<(NTOT + 255) / 256, 256, 0, stream>>>(Pos, PosP, pts);
  knn_kernel<<<dim3(NPTS_ / 256, BATCH), 256, 0, stream>>>(Pos, Nbr);
  basis_grad_kernel<<<(NTOT + 127) / 128, 128, 0, stream>>>(Pos, Nbr, Gc);
  grad_apply_kernel<<<((NTOT * 3) + 255) / 256, 256, 0, stream>>>(V, Pos, Gc, Nbr, 3);

  // ---- three DGAConv layers ----
  auto layer = [&](int l, const float* Xg /*padded, ld=ldx*/, const float* Xe /*ld=cin*/, int ldx) {
    const ConvIdx& C = convIdx[l];
    int cin = cinL[l], cout = coutL[l];
    int cen = (l == 0), vec = (l < 2);
    int ldxc = (l == 0) ? 32 : 4 * cin;            // padded xcat stride
    // x_max -> T2 (fused gather + GEMM + max; relu commutes with max)
    edge(Xg, ldx, nullptr, Wsmax[l], P(C.smax_b), nullptr, nullptr, T2, cin, cout,
         0, cen, 0, 1, 1.f, 0);
    // div / curl
    divop(D1, V, cin, 0);
    divop(D2, V, cin, 1);
    if (ldxc != 4 * cin) hipMemsetAsync(XC, 0, (size_t)NTOT * ldxc * 4, stream);
    cat_x_kernel<<<((NTOT * 4 * cin) + 255) / 256, 256, 0, stream>>>(XC, ldxc, Xe, D1, D2, V, cin);
    // x = x_max + relu(s_mlp(xcat))
    gemm(XC, ldxc, Wsmlp[l], P(C.smlp_b), nullptr, nullptr, T2, T1, NTOT, 4 * cin, cout, 1, 1.f);
    // attn (length-1 sequence -> two projections)
    gemm(T1, cout, Wasv[l], P(C.asv_b), nullptr, nullptr, nullptr, T2, NTOT, cout, cout, 0, 1.f);
    gemm(T2, cout, Waso[l], P(C.aso_b), nullptr, nullptr, nullptr, X,  NTOT, cout, cout, 0, 1.f);
    // scalar GLU (fused 4-channel) -> D1, then x = LN(glu + rowmax)
    glu(X, cout, Wgsv[l], Wgsg[l], P(C.gs_bv), P(C.gs_bg), P(C.gs_alpha), D1, NTOT, cout);
    ln_kernel<<<(NTOT + 255) / 256, 256, 0, stream>>>(X, D1, P(C.ln_g), P(C.ln_b), cout);
    if (vec) {
      divop(D1, V, cin, 0);
      grad_apply_kernel<<<((NTOT * cin) + 255) / 256, 256, 0, stream>>>(VT1, D1, Gc, Nbr, cin);
      divop(D2, V, cin, 1);
      grad_apply_kernel<<<((NTOT * cin) + 255) / 256, 256, 0, stream>>>(VT2, D2, Gc, Nbr, cin);
      grad_apply_kernel<<<((NTOT * cout) + 255) / 256, 256, 0, stream>>>(GX, X, Gc, Nbr, cout);
      int Kv = 2 * cin + cout;
      int ldv = (2 * Kv + 31) & ~31;               // 160 for layer 0, 512 for layer 1
      if (ldv != 2 * Kv) hipMemsetAsync(BIG, 0, (size_t)2 * NTOT * ldv * 4, stream);
      vcat_kernel<<<((NTOT * Kv) + 255) / 256, 256, 0, stream>>>(BIG, ldv, V, VT1, VT2, GX, cin, cout);
      gemm(BIG, ldv, Wvmlp[l], P(C.vmlp_b), nullptr, nullptr, nullptr, V, 2 * NTOT, 2 * Kv, cout, 1, 1.f);
      // vector attn
      gemm(V, cout, Wavv[l], P(C.avv_b), nullptr, nullptr, nullptr, VT1, 2 * NTOT, cout, cout, 0, 1.f);
      gemm(VT1, cout, Wavo[l], P(C.avo_b), nullptr, nullptr, nullptr, V, 2 * NTOT, cout, cout, 0, 1.f);
      // vector GLU -> GX; then v = glu + mean
      glu(V, cout, Wgvv[l], Wgvg[l], P(C.gv_bv), P(C.gv_bg), P(C.gv_alpha), GX, 2 * NTOT, cout);
      vmean_kernel<<<((2 * NTOT) + 255) / 256, 256, 0, stream>>>(V, GX, cout);
    }
    // layer-3 vector branch is dead code in the reference (v unused afterwards)
  };
  layer(0, PosP, Pos, 32);
  layer(1, X, X, 64);
  layer(2, X, X, 128);

  // ---- SE block ----
  se_reduce_kernel<<<2, 256, 0, stream>>>(SEs, X);
  se_mlp_kernel<<<1, 512, 0, stream>>>(SEf, SEs, P(91), P(90), P(87), P(86),
                                       P(93), P(92), P(89), P(88));
  se_apply_kernel<<<((NTOT * 128) + 255) / 256, 256, 0, stream>>>(T1, X, SEf);

  // ---- delta / post (fused f*kd GEMM + sum over k, BN+ReLU epilogue) ----
  const float bninv = 1.0f / sqrtf(1.0f + 1e-5f);
  hipMemsetAsync(KD3, 0, (size_t)EDG * 32 * 4, stream);
  kd3_kernel<<<(EDG + 255) / 256, 256, 0, stream>>>(KD3, Pos, Nbr);
  gemm(KD3, 32, Wdelta, P(80), P(79), P(78), nullptr, BIG, EDG, 3, 128, 1, bninv);
  edge(T1, 128, BIG, Wpost, P(84), P(83), P(82), (float*)d_out, 128, 128,
       1, 0, 1, 1, bninv, 1);
}
